// Interaction_85942295593201
// MI455X (gfx1250) — matrix operand order repaired and checked by reading, not compiled, not run on hardware
//
#include <hip/hip_runtime.h>

#define NN 25000
#define HH 64
#define RR 32
#define EE 400000

typedef __attribute__((ext_vector_type(16))) _Float16 v16h;
typedef __attribute__((ext_vector_type(8)))  float    v8f;

// ---------------------------------------------------------------------------
// WMMA helpers (gfx1250: V_WMMA_F32_16X16X32_F16, wave32)
// ---------------------------------------------------------------------------
__device__ __forceinline__ v8f wmma_f16(v16h a, v16h b, v8f c) {
    return __builtin_amdgcn_wmma_f32_16x16x32_f16(false, a, false, b,
                                                  (short)0, c, false, false);
}

// A fragment: 16x32 f16, row-major A[m*lda + k].
__device__ __forceinline__ v16h load_a_frag(const _Float16* A, int lda, int lane) {
    int m  = lane & 15;
    int kh = (lane >> 4) << 3;
    v16h a;
#pragma unroll
    for (int v = 0; v < 8; ++v) {
        int k = ((v < 4) ? 0 : 16) + kh + ((v & 3) << 1);
        a[2 * v]     = A[m * lda + k];
        a[2 * v + 1] = A[m * lda + k + 1];
    }
    return a;
}

// B fragment: 32x16 f16 where B[k][n] = W[n*ldw + k]  (computes act @ W^T).
__device__ __forceinline__ v16h load_b_frag(const _Float16* W, int ldw, int lane) {
    int n  = lane & 15;
    int kb = (lane >> 4) << 3;
    v16h b;
#pragma unroll
    for (int v = 0; v < 8; ++v) {
        int k = ((v < 4) ? 0 : 16) + kb + ((v & 3) << 1);
        b[2 * v]     = W[n * ldw + k];
        b[2 * v + 1] = W[n * ldw + k + 1];
    }
    return b;
}

// fast silu: x * rcp(1+exp(-x))  (v_exp_f32 + v_rcp_f32, no IEEE div sequence)
__device__ __forceinline__ float silu_f(float x) {
    return x * __builtin_amdgcn_rcpf(1.0f + __expf(-x));
}

// compact basis: d = {I, a01, a02, a12, s00, s11, s01, s02, s12}; s22 = -s00-s11
__device__ __forceinline__ void decompose9(const float T[9], float d[9]) {
    float I = (T[0] + T[4] + T[8]) * (1.0f / 3.0f);
    d[0] = I;
    d[1] = 0.5f * (T[1] - T[3]);
    d[2] = 0.5f * (T[2] - T[6]);
    d[3] = 0.5f * (T[5] - T[7]);
    d[4] = T[0] - I;
    d[5] = T[4] - I;
    d[6] = 0.5f * (T[1] + T[3]);
    d[7] = 0.5f * (T[2] + T[6]);
    d[8] = 0.5f * (T[5] + T[7]);
}

__device__ __forceinline__ void reconstruct9(const float d[9], float T[9]) {
    float I = d[0];
    T[0] = I + d[4];       T[4] = I + d[5];       T[8] = I - d[4] - d[5];
    T[1] = d[6] + d[1];    T[3] = d[6] - d[1];
    T[2] = d[7] + d[2];    T[6] = d[7] - d[2];
    T[5] = d[8] + d[3];    T[7] = d[8] - d[3];
}

__device__ __forceinline__ void mm3(const float* A, const float* B, float* C) {
#pragma unroll
    for (int r = 0; r < 3; ++r)
#pragma unroll
        for (int c = 0; c < 3; ++c)
            C[r * 3 + c] = A[r * 3 + 0] * B[c] + A[r * 3 + 1] * B[3 + c] + A[r * 3 + 2] * B[6 + c];
}

// ---------------------------------------------------------------------------
// K0: f32 -> f16 weight conversion
// ---------------------------------------------------------------------------
__global__ void k_f32_to_f16(const float* __restrict__ src, _Float16* __restrict__ dst, int n) {
    int i = blockIdx.x * blockDim.x + threadIdx.x;
    if (i < n) dst[i] = (_Float16)src[i];
}

// ---------------------------------------------------------------------------
// K1: normalize X, compact decomposition, zero Y accumulator
// ---------------------------------------------------------------------------
__global__ void k_node_prep(const float* __restrict__ X, float* __restrict__ Xn,
                            float* __restrict__ D, float* __restrict__ Yagg) {
    int i = blockIdx.x * blockDim.x + threadIdx.x;
    if (i >= NN * HH) return;
    const float* x = X + (size_t)i * 9;
    float t[9], nn = 0.0f;
#pragma unroll
    for (int c = 0; c < 9; ++c) { t[c] = x[c]; nn += t[c] * t[c]; }
    float inv = __builtin_amdgcn_rcpf(nn + 1.0f);
#pragma unroll
    for (int c = 0; c < 9; ++c) t[c] *= inv;
    float d[9];
    decompose9(t, d);
    float* xo = Xn + (size_t)i * 9;
    float* dp = D + (size_t)i * 9;
    float* yp = Yagg + (size_t)i * 9;
#pragma unroll
    for (int c = 0; c < 9; ++c) { xo[c] = t[c]; dp[c] = d[c]; yp[c] = 0.0f; }
}

// ---------------------------------------------------------------------------
// K2: edge MLP (3 WMMA GEMM layers + silu + cutoff) fused with message scatter
// block = 256 threads (8 waves), 32 edges per block
// ---------------------------------------------------------------------------
#define TE 32
__global__ void __launch_bounds__(256)
k_edge_mlp_scatter(const float* __restrict__ edge_attr,
                   const float* __restrict__ edge_weight,
                   const int*   __restrict__ edge_index,
                   const float* __restrict__ b1, const float* __restrict__ b2,
                   const float* __restrict__ b3,
                   const _Float16* __restrict__ Wh,   // W1h@0, W2h@2048, W3h@10240
                   const float* __restrict__ D,
                   float* __restrict__ Yagg)
{
    __shared__ _Float16 sW1[64 * 32];     //  4 KB
    __shared__ _Float16 sW2[128 * 64];    // 16 KB
    __shared__ _Float16 sA[TE * 32];      //  2 KB
    __shared__ _Float16 sH1[TE * 64];     //  4 KB
    __shared__ _Float16 sH2[TE * 128];    //  8 KB
    __shared__ float    sF[TE * 192];     // 24 KB
    __shared__ float    sC[TE];

    const int tid  = threadIdx.x;
    const int lane = tid & 31;
    const int wv   = tid >> 5;
    const int e0   = blockIdx.x * TE;

    for (int i = tid; i < 2048; i += 256) sW1[i] = Wh[i];
    for (int i = tid; i < 8192; i += 256) sW2[i] = Wh[2048 + i];
    for (int i = tid; i < TE * 32; i += 256)
        sA[i] = (_Float16)edge_attr[(size_t)(e0 + (i >> 5)) * RR + (i & 31)];
    if (tid < TE) {
        float w = edge_weight[e0 + tid];
        float c = 0.5f * (__cosf(w * 0.62831853071795864769f) + 1.0f);  // pi/5
        sC[tid] = (w < 5.0f) ? c : 0.0f;
    }
    __syncthreads();

    // ---- layer 1: [32,32] @ W1^T -> [32,64]; 2 M x 4 N tiles, 1 per wave ----
    {
        int mt = wv >> 2, nt = wv & 3;
        v16h a = load_a_frag(sA + mt * 16 * 32, 32, lane);
        v16h b = load_b_frag(sW1 + nt * 16 * 32, 32, lane);
        v8f acc = {};
        acc = wmma_f16(a, b, acc);
        int col = lane & 15;
#pragma unroll
        for (int v = 0; v < 8; ++v) {
            int row = mt * 16 + v + ((lane >> 4) << 3);
            int gc = nt * 16 + col;
            sH1[row * 64 + gc] = (_Float16)silu_f(acc[v] + b1[gc]);
        }
    }
    __syncthreads();

    // ---- layer 2: [32,64] @ W2^T -> [32,128]; 16 tiles, 2/wave, K=64 ----
    for (int t = wv; t < 16; t += 8) {
        int mt = t >> 3, nt = t & 7;
        v8f acc = {};
#pragma unroll
        for (int ks = 0; ks < 2; ++ks) {
            v16h a = load_a_frag(sH1 + mt * 16 * 64 + ks * 32, 64, lane);
            v16h b = load_b_frag(sW2 + nt * 16 * 64 + ks * 32, 64, lane);
            acc = wmma_f16(a, b, acc);
        }
        int col = lane & 15;
#pragma unroll
        for (int v = 0; v < 8; ++v) {
            int row = mt * 16 + v + ((lane >> 4) << 3);
            int gc = nt * 16 + col;
            sH2[row * 128 + gc] = (_Float16)silu_f(acc[v] + b2[gc]);
        }
    }
    __syncthreads();

    // ---- layer 3: [32,128] @ W3^T -> [32,192]; 24 tiles, 3/wave, K=128.
    // W3 fragments stream from global (L2-resident, 48 KB) to keep LDS small.
    const _Float16* W3h = Wh + 10240;
    for (int t = wv; t < 24; t += 8) {
        int mt = t / 12, nt = t % 12;
        v8f acc = {};
#pragma unroll
        for (int ks = 0; ks < 4; ++ks) {
            v16h a = load_a_frag(sH2 + mt * 16 * 128 + ks * 32, 128, lane);
            v16h b = load_b_frag(W3h + nt * 16 * 128 + ks * 32, 128, lane);
            acc = wmma_f16(a, b, acc);
        }
        int col = lane & 15;
#pragma unroll
        for (int v = 0; v < 8; ++v) {
            int row = mt * 16 + v + ((lane >> 4) << 3);
            int gc = nt * 16 + col;
            sF[row * 192 + gc] = silu_f(acc[v] + b3[gc]) * sC[row];
        }
    }
    __syncthreads();

    // ---- fused gather/scatter of compact messages: 32 edges x 64 channels ----
    for (int i = tid; i < TE * HH; i += 256) {
        int el = i >> 6;
        int h  = i & 63;
        int e  = e0 + el;
        int dst = edge_index[e];         // edge_index[0]
        int src = edge_index[EE + e];    // edge_index[1]
        float f0 = sF[el * 192 + h * 3 + 0];
        float f1 = sF[el * 192 + h * 3 + 1];
        float f2 = sF[el * 192 + h * 3 + 2];
        const float* ds = D + ((size_t)src * HH + h) * 9;
        float* yd = Yagg + ((size_t)dst * HH + h) * 9;
        atomicAdd(&yd[0], f0 * ds[0]);
        atomicAdd(&yd[1], f1 * ds[1]);
        atomicAdd(&yd[2], f1 * ds[2]);
        atomicAdd(&yd[3], f1 * ds[3]);
#pragma unroll
        for (int c = 4; c < 9; ++c) atomicAdd(&yd[c], f2 * ds[c]);
    }
}

// ---------------------------------------------------------------------------
// K3: _tensor_linear channel mixing on compact basis via WMMA
// out[:, :, c] = in[:, :, c] @ Wc^T ; Wc = WI (c=0) / WA (c=1..3) / WS (c=4..8)
// block = 256 threads (8 waves), 32 nodes per block
// ---------------------------------------------------------------------------
__global__ void __launch_bounds__(256)
k_channel_mix(const float* __restrict__ inC, float* __restrict__ outC,
              const _Float16* __restrict__ Wh, int wbase)
{
    __shared__ _Float16 sIn[9][32 * 64];   // 36 KB
    __shared__ _Float16 sW[3 * 64 * 64];   // 24 KB
    const int tid = threadIdx.x, lane = tid & 31, wv = tid >> 5;
    const int n0 = blockIdx.x * 32;

    for (int i = tid; i < 3 * 4096; i += 256) sW[i] = Wh[wbase + i];
    for (int i = tid; i < 32 * 64 * 9; i += 256) {
        int m = i / 576;
        int r = i - m * 576;
        int k = r / 9;
        int c = r - k * 9;
        int n = n0 + m;
        float v = (n < NN) ? inC[(size_t)n * 576 + k * 9 + c] : 0.0f;
        sIn[c][m * 64 + k] = (_Float16)v;
    }
    __syncthreads();

    // 9 comps x 2 M-tiles x 4 N-tiles = 72 output tiles, K=64 (2 WMMA each)
    for (int t = wv; t < 72; t += 8) {
        int c  = t >> 3;
        int r  = t & 7;
        int mt = r >> 2, nt = r & 3;
        int wsel = (c == 0) ? 0 : ((c < 4) ? 1 : 2);
        const _Float16* W = sW + wsel * 4096 + nt * 16 * 64;
        v8f acc = {};
#pragma unroll
        for (int ks = 0; ks < 2; ++ks) {
            v16h a = load_a_frag(&sIn[c][mt * 16 * 64 + ks * 32], 64, lane);
            v16h b = load_b_frag(W + ks * 32, 64, lane);
            acc = wmma_f16(a, b, acc);
        }
        int col = lane & 15;
#pragma unroll
        for (int v = 0; v < 8; ++v) {
            int row = mt * 16 + v + ((lane >> 4) << 3);
            int n = n0 + row;
            if (n < NN) outC[(size_t)n * 576 + (nt * 16 + col) * 9 + c] = acc[v];
        }
    }
}

// ---------------------------------------------------------------------------
// K4: Xnew = Y@Xin + Xin@Y ; normalize ; decompose
// ---------------------------------------------------------------------------
__global__ void k_node_combine(const float* __restrict__ XinC,
                               const float* __restrict__ YC,
                               float* __restrict__ XnewC)
{
    int i = blockIdx.x * blockDim.x + threadIdx.x;
    if (i >= NN * HH) return;
    float di[9], dy[9], Xin[9], Y[9], T1[9], T2[9], Xnew[9], dout[9];
#pragma unroll
    for (int c = 0; c < 9; ++c) { di[c] = XinC[(size_t)i * 9 + c]; dy[c] = YC[(size_t)i * 9 + c]; }
    reconstruct9(di, Xin);
    reconstruct9(dy, Y);
    mm3(Y, Xin, T1);    // Am
    mm3(Xin, Y, T2);    // Bm
    float nn = 0.0f;
#pragma unroll
    for (int c = 0; c < 9; ++c) { Xnew[c] = T1[c] + T2[c]; nn += Xnew[c] * Xnew[c]; }
    float inv = __builtin_amdgcn_rcpf(nn + 1.0f);
#pragma unroll
    for (int c = 0; c < 9; ++c) Xnew[c] *= inv;
    decompose9(Xnew, dout);
#pragma unroll
    for (int c = 0; c < 9; ++c) XnewC[(size_t)i * 9 + c] = dout[c];
}

// ---------------------------------------------------------------------------
// K5: out = Xn + (dX + dX@dX) * (1 + 0.1 q)
// ---------------------------------------------------------------------------
__global__ void k_node_out(const float* __restrict__ dXC, const float* __restrict__ Xn,
                           const float* __restrict__ q, float* __restrict__ out)
{
    int i = blockIdx.x * blockDim.x + threadIdx.x;
    if (i >= NN * HH) return;
    int n = i >> 6;
    float d[9], dX[9], M[9];
#pragma unroll
    for (int c = 0; c < 9; ++c) d[c] = dXC[(size_t)i * 9 + c];
    reconstruct9(d, dX);
    mm3(dX, dX, M);
    float cf = 1.0f + 0.1f * q[n];
#pragma unroll
    for (int c = 0; c < 9; ++c)
        out[(size_t)i * 9 + c] = Xn[(size_t)i * 9 + c] + (dX[c] + M[c]) * cf;
}

// ---------------------------------------------------------------------------
extern "C" void kernel_launch(void* const* d_in, const int* in_sizes, int n_in,
                              void* d_out, int out_size, void* d_ws, size_t ws_size,
                              hipStream_t stream) {
    const float* X   = (const float*)d_in[0];
    const int*   ei  = (const int*)d_in[1];
    const float* ew  = (const float*)d_in[2];
    const float* ea  = (const float*)d_in[3];
    const float* q   = (const float*)d_in[4];
    const float* W1  = (const float*)d_in[5];
    const float* b1  = (const float*)d_in[6];
    const float* W2  = (const float*)d_in[7];
    const float* b2  = (const float*)d_in[8];
    const float* W3  = (const float*)d_in[9];
    const float* b3  = (const float*)d_in[10];
    const float* WIi = (const float*)d_in[11];
    const float* WAi = (const float*)d_in[12];
    const float* WSi = (const float*)d_in[13];
    const float* WIo = (const float*)d_in[14];
    const float* WAo = (const float*)d_in[15];
    const float* WSo = (const float*)d_in[16];

    const size_t NH9 = (size_t)NN * HH * 9;
    float* Xn   = (float*)d_ws;
    float* D    = Xn + NH9;        // later reused as XnewC
    float* Yagg = D + NH9;         // later reused as dXC
    float* XinC = Yagg + NH9;
    _Float16* Wh = (_Float16*)(XinC + NH9);

    auto cvt = [&](const float* s, int off, int n) {
        k_f32_to_f16<<<(n + 255) / 256, 256, 0, stream>>>(s, Wh + off, n);
    };
    cvt(W1, 0, 2048);      cvt(W2, 2048, 8192);   cvt(W3, 10240, 24576);
    cvt(WIi, 34816, 4096); cvt(WAi, 38912, 4096); cvt(WSi, 43008, 4096);
    cvt(WIo, 47104, 4096); cvt(WAo, 51200, 4096); cvt(WSo, 55296, 4096);

    const int nh = NN * HH;
    k_node_prep<<<(nh + 255) / 256, 256, 0, stream>>>(X, Xn, D, Yagg);
    k_edge_mlp_scatter<<<EE / TE, 256, 0, stream>>>(ea, ew, ei, b1, b2, b3, Wh, D, Yagg);
    k_channel_mix<<<(NN + 31) / 32, 256, 0, stream>>>(D, XinC, Wh, 34816);
    k_node_combine<<<(nh + 255) / 256, 256, 0, stream>>>(XinC, Yagg, D);          // XnewC -> D
    k_channel_mix<<<(NN + 31) / 32, 256, 0, stream>>>(D, Yagg, Wh, 47104);        // dXC -> Yagg
    k_node_out<<<(nh + 255) / 256, 256, 0, stream>>>(Yagg, Xn, q, (float*)d_out);
}
